// AttnDecoder_22771916603801
// MI455X (gfx1250) — matrix operand order repaired and checked by reading, not compiled, not run on hardware
//
#include <hip/hip_runtime.h>
#include <hip/hip_bf16.h>

// ---------------------------------------------------------------------------
// Problem constants (from reference)
// ---------------------------------------------------------------------------
#define BATCH   512
#define HDIM    256
#define LDIM    256
#define TSTEPS  64
#define VDIM    128
#define NLAYERS 2

#define TILE     16         // batch rows per workgroup (WMMA M)
#define NTHREADS 512        // 16 wave32 per workgroup
#define NWAVES   16
#define PITCH    260        // LDS row pitch for 256-wide fp32 rows (+4 pad)
#define GP       772        // LDS row pitch for 768-wide GRU gate rows
#define VP       132        // LDS row pitch for 128-wide logit rows

// Swizzled-weight tile sizes (in ushort/bf16 elements). One WMMA B tile
// (K=32 x N=16 bf16) = 32 lanes * 16 elems = 512 elements.
#define ATTN_SW_ELEMS  (16 * 16 * 512)   // N=256 (16 nt) K=512 (16 kt)
#define COMB_SW_ELEMS  (16 * 16 * 512)
#define GRU_SW_ELEMS   (48 *  8 * 512)   // N=768 (48 nt) K=256 (8 kt) per matrix
#define OUT_SW_ELEMS   ( 8 *  8 * 512)   // N=128 K=256

// Activation buffers in LDS: bf16, A-fragment swizzled, 16x256 -> 4096 elems.
#define ACT_ELEMS  4096                   // = 8 k-tiles * 512
#define ACT_BYTES  (ACT_ELEMS * 2)

// LDS plan: 5 bf16 activation buffers + fp32 sc/gi/gh/lg + toks
#define SMEM_BYTES (5 * ACT_BYTES + \
                    (TILE * PITCH + 2 * TILE * GP + TILE * VP + 16) * 4)

typedef __attribute__((ext_vector_type(16))) __bf16 v16bf;
typedef __attribute__((ext_vector_type(8)))  float  v8f;

union Frag { v16bf v; uint4 q[2]; unsigned short u[16]; };

// float -> bf16 bits, round-to-nearest-even
static __device__ __forceinline__ unsigned short f2bfbits(float f) {
    union { float f; unsigned u; } a; a.f = f;
    unsigned u = a.u;
    unsigned r = (u + 0x7FFFu + ((u >> 16) & 1u)) >> 16;
    return (unsigned short)r;
}
static __device__ __forceinline__ float bf2f(unsigned short s) {
    union { unsigned u; float f; } a; a.u = ((unsigned)s) << 16;
    return a.f;
}

// Swizzled index of element (m, k) inside an A-fragment-ordered buffer.
// ISA 05_wmma 16-bit A 16x32 layout: lane = m + 16*hi, elem i -> K =
// kt*32 + (i&7) + 8*hi + 16*(i>>3).  Inverse:
static __device__ __forceinline__ int swz_idx(int m, int k) {
    int kt   = k >> 5;
    int kin  = k & 31;
    int lane = (m & 15) + 16 * ((kin >> 3) & 1);
    int i    = (kin & 7) + 8 * (kin >> 4);
    return kt * 512 + lane * 16 + i;
}

// ---------------------------------------------------------------------------
// Prep kernel: swizzle fp32 weight W[N][K] (row-major, used as x @ W.T)
// into bf16 WMMA B-fragment tiles.  Tile order (nt, kt), kt contiguous.
// ISA B 32x16 layout: lane -> n = nt*16 + (lane&15); elem i -> K = kt*32 +
// (i&7) + 8*(lane>>4) + 16*(i>>3).
// ---------------------------------------------------------------------------
__global__ void swizzleB_kernel(const float* __restrict__ W,
                                unsigned short* __restrict__ out,
                                int N, int K) {
    int tid = blockIdx.x * blockDim.x + threadIdx.x;
    int total = N * K;
    if (tid >= total) return;
    int Kt     = K >> 5;
    int tile   = tid >> 9;
    int within = tid & 511;
    int lane   = within >> 4;
    int i      = within & 15;
    int nt = tile / Kt;
    int kt = tile - nt * Kt;
    int n  = nt * 16 + (lane & 15);
    int hi = lane >> 4;
    int k  = kt * 32 + (i & 7) + 8 * hi + 16 * (i >> 3);
    out[tid] = f2bfbits(W[(size_t)n * K + k]);
}

// ---------------------------------------------------------------------------
// 16xN GEMM with bf16 WMMA.  A in LDS in fragment order (2x ds_load_b128),
// B streamed from pre-swizzled global (2x global_load_b128).  Fully
// unrolled straight-line body (the backend scheduler pipelines it); NO
// runtime tile guard: every wave owns exactly NTQ tiles (the out-GEMM
// passes wave&7 so high waves duplicate identical work, keeping EXEC
// all-ones for every WMMA as the ISA requires).
//   CONCAT: A = [A0 | A1] along K (KT must be 16)
//   MODE 0: C fp32 natural (pitch CP) + bias
//   MODE 1: relu(C+bias) -> Csw bf16 swizzled (feeds next GEMM as A)
// ---------------------------------------------------------------------------
template<int KT, int NTQ, int MODE, bool CONCAT>
static __device__ __forceinline__ void
gemm16(const unsigned short* __restrict__ A0,
       const unsigned short* __restrict__ A1,
       const unsigned short* __restrict__ Bsw,
       float* __restrict__ C, int CP,
       unsigned short* __restrict__ Csw,
       const float* __restrict__ bias,
       int lane, int wv) {
    const int ntbase = wv * NTQ;
    v8f acc[NTQ] = {};
#pragma unroll
    for (int kt = 0; kt < KT; ++kt) {
        const unsigned short* asrc =
            (CONCAT && kt >= KT / 2) ? (A1 + (kt - KT / 2) * 512)
                                     : (A0 + kt * 512);
        Frag a;
        const uint4* ap = (const uint4*)(asrc + lane * 16);
        a.q[0] = ap[0];
        a.q[1] = ap[1];
        // pull the next K tile toward L0/L2 (global_prefetch_b8); speculative
        __builtin_prefetch((const void*)(Bsw +
            ((size_t)ntbase * KT + kt + 1) * 512 + lane * 16), 0, 1);
#pragma unroll
        for (int q = 0; q < NTQ; ++q) {
            const uint4* bp = (const uint4*)(Bsw +
                ((size_t)(ntbase + q) * KT + kt) * 512 + lane * 16);
            Frag b;
            b.q[0] = bp[0];
            b.q[1] = bp[1];
            acc[q] = __builtin_amdgcn_wmma_f32_16x16x32_bf16(
                false, a.v, false, b.v, (short)0, acc[q], false, false);
        }
    }

    const int nloc = lane & 15;
    const int hi   = lane >> 4;
#pragma unroll
    for (int q = 0; q < NTQ; ++q) {
        int n = (ntbase + q) * 16 + nloc;
        float bv = bias[n];
#pragma unroll
        for (int j = 0; j < 8; ++j) {
            int m = j + 8 * hi;            // ISA C layout: VGPR j -> M=j / j+8
            float v = acc[q][j] + bv;
            if (MODE == 1) {
                if (v < 0.0f) v = 0.0f;
                Csw[swz_idx(m, n)] = f2bfbits(v);
            } else {
                C[m * CP + n] = v;
            }
        }
    }
}

// wave-level butterfly reductions (row == wave, 32 lanes per row)
static __device__ __forceinline__ float wave_max(float v) {
#pragma unroll
    for (int off = 16; off >= 1; off >>= 1)
        v = fmaxf(v, __shfl_xor(v, off, 32));
    return v;
}
static __device__ __forceinline__ float wave_sum(float v) {
#pragma unroll
    for (int off = 16; off >= 1; off >>= 1)
        v += __shfl_xor(v, off, 32);
    return v;
}

// ---------------------------------------------------------------------------
// Main persistent decoder kernel: one workgroup owns 16 batch rows for all
// 64 time steps.  512 threads = 16 wave32; wave w handles reduction row w.
// ---------------------------------------------------------------------------
__global__ void __launch_bounds__(NTHREADS, 1)
attn_decoder_kernel(const float* __restrict__ x,          // [B,H,1,L]
                    const int*   __restrict__ y,          // [B,T]
                    const float* __restrict__ emb,        // [V,H]
                    const unsigned short* __restrict__ attn_sw,
                    const float* __restrict__ attn_b,     // [L]
                    const unsigned short* __restrict__ comb_sw,
                    const float* __restrict__ comb_b,     // [H]
                    const unsigned short* __restrict__ ih_sw,  // 2 layers
                    const unsigned short* __restrict__ hh_sw,  // 2 layers
                    const float* __restrict__ b_ih,       // [2,768]
                    const float* __restrict__ b_hh,       // [2,768]
                    const unsigned short* __restrict__ out_sw,
                    const float* __restrict__ out_b,      // [V]
                    float* __restrict__ outs,             // [B,V,T]
                    float* __restrict__ attns)            // [B,T,L]
{
    extern __shared__ char smemraw[];
    unsigned short* e_sw  = (unsigned short*)smemraw;            // bf16 swizzled
    unsigned short* ap_sw = e_sw  + ACT_ELEMS;
    unsigned short* g_sw  = ap_sw + ACT_ELEMS;
    unsigned short* h0_sw = g_sw  + ACT_ELEMS;
    unsigned short* h1_sw = h0_sw + ACT_ELEMS;
    float* sc  = (float*)(smemraw + 5 * ACT_BYTES);  // [16][PITCH] scores/aw
    float* gi  = sc  + TILE * PITCH;                 // [16][GP]
    float* gh  = gi  + TILE * GP;                    // [16][GP]
    float* lg  = gh  + TILE * GP;                    // [16][VP]
    int*  toks = (int*)(lg + TILE * VP);             // [16]

    const int tid  = threadIdx.x;
    const int lane = tid & 31;
    const int wave = tid >> 5;          // 0..15; also the reduction row
    const int b0   = blockIdx.x * TILE;

    // zero hidden state (bf16 zero == 0x0000)
    for (int i = tid; i < ACT_ELEMS; i += NTHREADS) { h0_sw[i] = 0; h1_sw[i] = 0; }
    __syncthreads();

    for (int t = 0; t < TSTEPS; ++t) {
        // ---- 1. teacher-forced tokens (SOS=0 at t=0) -----------------------
        if (tid < TILE)
            toks[tid] = (t == 0) ? 0 : y[(size_t)(b0 + tid) * TSTEPS + (t - 1)];
        __syncthreads();

        // ---- 2. embedding gather straight into swizzled bf16 ---------------
        for (int idx = tid; idx < ACT_ELEMS; idx += NTHREADS) {
            int kt = idx >> 9, rem = idx & 511, ln = rem >> 4, i = rem & 15;
            int m  = ln & 15, hi = ln >> 4;
            int k  = kt * 32 + (i & 7) + 8 * hi + 16 * (i >> 3);
            e_sw[idx] = f2bfbits(emb[(size_t)toks[m] * HDIM + k]);
        }
        __syncthreads();

        // ---- 3. attention scores: [e | h0] @ attn_w^T -> sc (fp32) ---------
        gemm16<16, 1, 0, true>(e_sw, h0_sw, attn_sw, sc, PITCH, nullptr,
                               attn_b, lane, wave);
        __syncthreads();

        // ---- 4. row softmax over L=256 (wave == row), write attns ----------
        {
            float mx = -3.4e38f;
            for (int j = lane; j < LDIM; j += 32)
                mx = fmaxf(mx, sc[wave * PITCH + j]);
            mx = wave_max(mx);
            float s = 0.0f;
            float ev[LDIM / 32];
#pragma unroll
            for (int u = 0; u < LDIM / 32; ++u) {
                ev[u] = __expf(sc[wave * PITCH + u * 32 + lane] - mx);
                s += ev[u];
            }
            float inv = 1.0f / wave_sum(s);
#pragma unroll
            for (int u = 0; u < LDIM / 32; ++u) {
                float p = ev[u] * inv;
                sc[wave * PITCH + u * 32 + lane] = p;
                __builtin_nontemporal_store(
                    p,
                    &attns[((size_t)(b0 + wave) * TSTEPS + t) * LDIM + u * 32 + lane]);
            }
        }
        __syncthreads();

        // ---- 5. attn apply: ap[r][h] = sum_l aw[r][l] * x[b0+r][h][0][l] ---
        // x rows contiguous in l and L2-resident (134 MB < 192 MB L2).
        for (int i = tid; i < TILE * HDIM; i += NTHREADS) {
            int r = i >> 8, hc = i & 255;
            const float4* xr4 =
                (const float4*)(x + ((size_t)(b0 + r) * HDIM + hc) * LDIM);
            const float4* aw4 = (const float4*)(sc + r * PITCH);
            float s = 0.0f;
#pragma unroll 4
            for (int l4 = 0; l4 < LDIM / 4; ++l4) {
                float4 xv = xr4[l4];
                float4 av = aw4[l4];
                s += av.x * xv.x + av.y * xv.y + av.z * xv.z + av.w * xv.w;
            }
            ap_sw[swz_idx(r, hc)] = f2bfbits(s);
        }
        __syncthreads();

        // ---- 6. combine: g = relu([ap|e] @ comb_w^T + b) -> swizzled bf16 --
        gemm16<16, 1, 1, true>(ap_sw, e_sw, comb_sw, nullptr, 0, g_sw,
                               comb_b, lane, wave);
        __syncthreads();

        // ---- 7. two GRU layers --------------------------------------------
        for (int l = 0; l < NLAYERS; ++l) {
            const unsigned short* xin = (l == 0) ? g_sw  : h0_sw;
            unsigned short*       hp  = (l == 0) ? h0_sw : h1_sw;
            gemm16<8, 3, 0, false>(xin, nullptr,
                                   ih_sw + (size_t)l * GRU_SW_ELEMS,
                                   gi, GP, nullptr, b_ih + l * 768, lane, wave);
            gemm16<8, 3, 0, false>(hp, nullptr,
                                   hh_sw + (size_t)l * GRU_SW_ELEMS,
                                   gh, GP, nullptr, b_hh + l * 768, lane, wave);
            __syncthreads();
            // fused GRU gates, updating hidden in swizzled bf16 order
            for (int idx = tid; idx < ACT_ELEMS; idx += NTHREADS) {
                int kt = idx >> 9, rem = idx & 511, ln = rem >> 4, i = rem & 15;
                int m  = ln & 15, hi = ln >> 4;
                int k  = kt * 32 + (i & 7) + 8 * hi + 16 * (i >> 3);
                float ir = gi[m * GP + k];
                float iz = gi[m * GP + 256 + k];
                float in_ = gi[m * GP + 512 + k];
                float hr = gh[m * GP + k];
                float hz = gh[m * GP + 256 + k];
                float hn = gh[m * GP + 512 + k];
                float hprev = bf2f(hp[idx]);
                float rg = 1.0f / (1.0f + __expf(-(ir + hr)));
                float zg = 1.0f / (1.0f + __expf(-(iz + hz)));
                float ng = tanhf(in_ + rg * hn);
                hp[idx] = f2bfbits((1.0f - zg) * ng + zg * hprev);
            }
            __syncthreads();
        }

        // ---- 8. logits: h1 @ out_w^T + out_b -> lg (16x128 fp32) -----------
        // 8 N tiles, 16 waves: waves 8..15 duplicate waves 0..7 (identical
        // values, benign duplicate LDS stores) so EXEC stays all-ones.
        gemm16<8, 1, 0, false>(h1_sw, nullptr, out_sw, lg, VP, nullptr,
                               out_b, lane, wave & 7);
        __syncthreads();

        // ---- 9. log_softmax over V=128 (wave == row), write outs [B,V,T] ---
        {
            float mx = -3.4e38f;
#pragma unroll
            for (int u = 0; u < VDIM / 32; ++u)
                mx = fmaxf(mx, lg[wave * VP + u * 32 + lane]);
            mx = wave_max(mx);
            float s = 0.0f;
#pragma unroll
            for (int u = 0; u < VDIM / 32; ++u)
                s += __expf(lg[wave * VP + u * 32 + lane] - mx);
            float lse = mx + __logf(wave_sum(s));
#pragma unroll
            for (int u = 0; u < VDIM / 32; ++u)
                __builtin_nontemporal_store(
                    lg[wave * VP + u * 32 + lane] - lse,
                    &outs[((size_t)(b0 + wave) * VDIM + u * 32 + lane) * TSTEPS + t]);
        }
        __syncthreads();
    }
}

// ---------------------------------------------------------------------------
// Host launcher
// ---------------------------------------------------------------------------
static inline void launch_swizzle(const float* W, unsigned short* o,
                                  int N, int K, hipStream_t stream) {
    int total = N * K;
    swizzleB_kernel<<<(total + 255) / 256, 256, 0, stream>>>(W, o, N, K);
}

extern "C" void kernel_launch(void* const* d_in, const int* in_sizes, int n_in,
                              void* d_out, int out_size, void* d_ws, size_t ws_size,
                              hipStream_t stream) {
    (void)in_sizes; (void)n_in; (void)out_size; (void)ws_size;

    const float* x      = (const float*)d_in[0];
    const int*   y      = (const int*)  d_in[1];
    const float* emb    = (const float*)d_in[2];
    const float* attn_w = (const float*)d_in[3];
    const float* attn_b = (const float*)d_in[4];
    const float* comb_w = (const float*)d_in[5];
    const float* comb_b = (const float*)d_in[6];
    const float* w_ih   = (const float*)d_in[7];
    const float* w_hh   = (const float*)d_in[8];
    const float* b_ih   = (const float*)d_in[9];
    const float* b_hh   = (const float*)d_in[10];
    const float* out_w  = (const float*)d_in[11];
    const float* out_b  = (const float*)d_in[12];

    unsigned short* ws      = (unsigned short*)d_ws;
    unsigned short* attn_sw = ws;
    unsigned short* comb_sw = attn_sw + ATTN_SW_ELEMS;
    unsigned short* ih_sw   = comb_sw + COMB_SW_ELEMS;
    unsigned short* hh_sw   = ih_sw   + 2 * GRU_SW_ELEMS;
    unsigned short* out_sw  = hh_sw   + 2 * GRU_SW_ELEMS;

    // Deterministic per-call weight bf16 swizzle (tiny: ~4 MB read total).
    launch_swizzle(attn_w,           attn_sw,              256, 512, stream);
    launch_swizzle(comb_w,           comb_sw,              256, 512, stream);
    launch_swizzle(w_ih,             ih_sw,                768, 256, stream);
    launch_swizzle(w_ih + 768 * 256, ih_sw + GRU_SW_ELEMS, 768, 256, stream);
    launch_swizzle(w_hh,             hh_sw,                768, 256, stream);
    launch_swizzle(w_hh + 768 * 256, hh_sw + GRU_SW_ELEMS, 768, 256, stream);
    launch_swizzle(out_w,            out_sw,               128, 256, stream);

    float* outs  = (float*)d_out;                              // [B,V,T]
    float* attns = outs + (size_t)BATCH * VDIM * TSTEPS;       // [B,T,L]

    attn_decoder_kernel<<<dim3(BATCH / TILE), dim3(NTHREADS), SMEM_BYTES,
                          stream>>>(
        x, y, emb, attn_sw, attn_b, comb_sw, comb_b,
        ih_sw, hh_sw, b_ih, b_hh, out_sw, out_b, outs, attns);
}
